// DGCNNSiteEmbed_39298950758860
// MI455X (gfx1250) — matrix operand order repaired and checked by reading, not compiled, not run on hardware
//
#include <hip/hip_runtime.h>
#include <hip/hip_bf16.h>

// ---------------------------------------------------------------------------
// Types for CDNA5 WMMA (gfx1250, wave32)
// ---------------------------------------------------------------------------
typedef __bf16 bf16_t;
typedef __attribute__((ext_vector_type(16))) __bf16 v16bf;
typedef __attribute__((ext_vector_type(8)))  __bf16 v8bf;
typedef __attribute__((ext_vector_type(8)))  float  v8f;

union BV { v16bf v; v8bf h[2]; };

__device__ __forceinline__ v8f wmma_bf16(v16bf a, v16bf b, v8f c) {
  // D = A(16x32 bf16) * B(32x16 bf16) + C(16x16 f32)
  return __builtin_amdgcn_wmma_f32_16x16x32_bf16(false, a, false, b,
                                                 (short)0, c, false, false);
}

#define NPTS 8192
#define KNN  40
#define BIGF 3.0e38f

// ---------------------------------------------------------------------------
// Pack a [Krows x Ncols] f32 weight into WMMA B-operand order (bf16).
// Layout: Wp[((nt*KT + kt)*32 + lane)*16 + h] = W[kt*32 + (h/8)*16 + (lane/16)*8 + h%8][nt*16 + lane%16]
// Rows >= Krows are zero (K padding).
// ---------------------------------------------------------------------------
__global__ void pack_w_kernel(const float* __restrict__ W, int Krows, int Ncols,
                              int KT, int total, bf16_t* __restrict__ Wp) {
  int tid = blockIdx.x * blockDim.x + threadIdx.x;
  if (tid >= total) return;
  int h    = tid & 15;
  int lane = (tid >> 4) & 31;
  int blk  = tid >> 9;
  int kt   = blk % KT;
  int nt   = blk / KT;
  int k = kt * 32 + ((h >> 3) << 4) + ((lane >> 4) << 3) + (h & 7);
  int n = nt * 16 + (lane & 15);
  float v = (k < Krows) ? W[(size_t)k * Ncols + n] : 0.0f;
  Wp[tid] = (bf16_t)v;
}

// ---------------------------------------------------------------------------
// f32 [M x cols] (row stride ldsrc) -> bf16 [M x 128] zero-padded (+ sq norms)
// ---------------------------------------------------------------------------
__global__ void cvt_pad_kernel(const float* __restrict__ src, int cols, int ldsrc,
                               bf16_t* __restrict__ dst, float* __restrict__ sq,
                               int M) {
  int n = blockIdx.x * blockDim.x + threadIdx.x;
  if (n >= M) return;
  float acc = 0.0f;
  for (int c = 0; c < 128; ++c) {
    float v = (c < cols) ? src[(size_t)n * ldsrc + c] : 0.0f;
    dst[(size_t)n * 128 + c] = (bf16_t)v;
    acc += v * v;
  }
  if (sq) sq[n] = acc;
}

// ---------------------------------------------------------------------------
// kNN top-40: 16 queries per 256-thread block (8 waves).  Gram tile G[16x128]
// per candidate step via WMMA (each wave owns a 16x16 subtile), then
// d = |xi|^2 + |xj|^2 - 2G, merged into per-query sorted top-40 lists in LDS.
// KT templated so A register array indexing is static (no scratch spill).
// ---------------------------------------------------------------------------
template <int KT>
__global__ __launch_bounds__(256) void knn_topk_kernel(
    const bf16_t* __restrict__ Xb, const float* __restrict__ sq,
    const int* __restrict__ batch, int* __restrict__ out_idx) {
  __shared__ __align__(16) float dtile[16 * 128];
  __shared__ float topd[16 * KNN];
  __shared__ int   topi[16 * KNN];

  const int tid = threadIdx.x, lane = tid & 31, wid = tid >> 5;
  const int q0 = blockIdx.x * 16;

  for (int j = tid; j < 16 * KNN; j += 256) { topd[j] = BIGF; topi[j] = 0; }

  // A operand: query rows, resident for whole kernel
  BV a[KT];
  {
    int row = q0 + (lane & 15);
    const bf16_t* base = Xb + (size_t)row * 128 + ((lane >> 4) << 4);
#pragma unroll
    for (int kt = 0; kt < KT; ++kt) {
      a[kt].h[0] = *(const v8bf*)(base + kt * 32);
      a[kt].h[1] = *(const v8bf*)(base + kt * 32 + 8);
    }
  }
  __syncthreads();

  for (int ct = 0; ct < NPTS / 128; ++ct) {
    const int cb = ct * 128;
    const int cand = cb + wid * 16 + (lane & 15);
    const bf16_t* bbase = Xb + (size_t)cand * 128 + ((lane >> 4) << 4);
    v8f acc = {};
#pragma unroll
    for (int kt = 0; kt < KT; ++kt) {
      v16bf bv = *(const v16bf*)(bbase + kt * 32);
      acc = wmma_bf16(a[kt].v, bv, acc);
    }
    float sqc = sq[cand];
    int   bc  = batch[cand];
#pragma unroll
    for (int v = 0; v < 8; ++v) {
      int qr = v + ((lane >> 4) << 3);
      float d = sq[q0 + qr] + sqc - 2.0f * acc[v];
      if (batch[q0 + qr] != bc) d = BIGF;
      dtile[qr * 128 + wid * 16 + (lane & 15)] = d;
    }
    __syncthreads();
    if (tid < 16) {
      const int q = tid;
      float* td = &topd[q * KNN];
      int*   ti = &topi[q * KNN];
      float worst = td[KNN - 1];
      for (int c = 0; c < 128; ++c) {
        float d = dtile[q * 128 + c];
        if (d < worst) {
          int pos = KNN - 1;
          while (pos > 0 && td[pos - 1] > d) {
            td[pos] = td[pos - 1]; ti[pos] = ti[pos - 1]; --pos;
          }
          td[pos] = d; ti[pos] = cb + c;
          worst = td[KNN - 1];
        }
      }
    }
    __syncthreads();
  }
  for (int j = tid; j < 16 * KNN; j += 256) {
    int q = j / KNN, s = j % KNN;
    out_idx[(size_t)(q0 + q) * KNN + s] = topi[j];
  }
}

// ---------------------------------------------------------------------------
// TNet EdgeConv (two MLP layers fused): 2 points/block (M = 80 edge rows).
// E1[80x32] (xi|xj-xi, 6 real cols) -> WMMA -> H1[80x64] bf16 ->
// WMMA -> H2[80x128] f32 -> max over 40 edges -> out[N,128].
// ---------------------------------------------------------------------------
__global__ __launch_bounds__(256) void tnet_edgeconv_kernel(
    const float* __restrict__ xyz, const int* __restrict__ nbr,
    const bf16_t* __restrict__ W1p, const float* __restrict__ b1,
    const float* __restrict__ s1, const float* __restrict__ t1,
    const bf16_t* __restrict__ W2p, const float* __restrict__ b2,
    const float* __restrict__ s2, const float* __restrict__ t2,
    float* __restrict__ out) {
  __shared__ __align__(16) bf16_t E1[80 * 32];
  __shared__ __align__(16) bf16_t H1[80 * 64];
  __shared__ __align__(16) float  H2[80 * 128];
  const int tid = threadIdx.x, lane = tid & 31, wid = tid >> 5;
  const int p0 = blockIdx.x * 2;

  for (int r = wid; r < 80; r += 8) {
    int p = r / KNN, e = r % KNN;
    int i = p0 + p, j = nbr[(size_t)i * KNN + e];
    if (lane < 3) {
      float xi = xyz[(size_t)i * 3 + lane];
      float xj = xyz[(size_t)j * 3 + lane];
      E1[r * 32 + lane]     = (bf16_t)xi;
      E1[r * 32 + 3 + lane] = (bf16_t)(xj - xi);
    }
    if (lane >= 6) E1[r * 32 + lane] = (bf16_t)0.0f;
  }
  __syncthreads();

  // GEMM1: 5 mtiles x 4 ntiles, KT = 1
  for (int tI = wid; tI < 20; tI += 8) {
    int mt = tI >> 2, nt = tI & 3;
    int row = mt * 16 + (lane & 15);
    const bf16_t* abase = E1 + row * 32 + ((lane >> 4) << 3);
    BV av;
    av.h[0] = *(const v8bf*)(abase);
    av.h[1] = *(const v8bf*)(abase + 16);
    v16bf bv = *(const v16bf*)(W1p + nt * 512 + lane * 16);
    v8f c = {};
    c = wmma_bf16(av.v, bv, c);
    int coln = nt * 16 + (lane & 15);
    float b = b1[coln], s = s1[coln], t = t1[coln];
#pragma unroll
    for (int v = 0; v < 8; ++v) {
      int r = mt * 16 + v + ((lane >> 4) << 3);
      float val = fmaxf(c[v] + b, 0.0f) * s + t;
      H1[r * 64 + coln] = (bf16_t)val;
    }
  }
  __syncthreads();

  // GEMM2: 5 mtiles x 8 ntiles, KT = 2
  for (int tI = wid; tI < 40; tI += 8) {
    int mt = tI >> 3, nt = tI & 7;
    int row = mt * 16 + (lane & 15);
    const bf16_t* abase = H1 + row * 64 + ((lane >> 4) << 3);
    v8f c = {};
#pragma unroll
    for (int kt = 0; kt < 2; ++kt) {
      BV av;
      av.h[0] = *(const v8bf*)(abase + kt * 32);
      av.h[1] = *(const v8bf*)(abase + kt * 32 + 16);
      v16bf bv = *(const v16bf*)(W2p + (nt * 2 + kt) * 512 + lane * 16);
      c = wmma_bf16(av.v, bv, c);
    }
    int coln = nt * 16 + (lane & 15);
    float b = b2[coln], s = s2[coln], t = t2[coln];
#pragma unroll
    for (int v = 0; v < 8; ++v) {
      int r = mt * 16 + v + ((lane >> 4) << 3);
      H2[r * 128 + coln] = fmaxf(c[v] + b, 0.0f) * s + t;
    }
  }
  __syncthreads();

  {
    int p = tid >> 7, c = tid & 127;
    float m = -BIGF;
    for (int e = 0; e < KNN; ++e) m = fmaxf(m, H2[(p * KNN + e) * 128 + c]);
    out[(size_t)(p0 + p) * 128 + c] = m;
  }
}

// ---------------------------------------------------------------------------
// Fused single-layer EdgeConv: 2 points/block, M = 80 edge rows, NOUT = 128.
// E[80 x KPAD] bf16 (union with H[80 x 128] f32) in 40 KB LDS.
// Wave w owns output-column tile w; accumulators for all 5 mtiles held in
// registers across the E->H buffer reuse barrier.
// ---------------------------------------------------------------------------
template <int CIN, int KPAD>
__global__ __launch_bounds__(256) void edgeconv_kernel(
    const bf16_t* __restrict__ Xb, const int* __restrict__ nbr,
    const bf16_t* __restrict__ Wp, const float* __restrict__ bias,
    const float* __restrict__ sc, const float* __restrict__ sh,
    float* __restrict__ out) {
  constexpr int M  = 80;
  constexpr int KT = KPAD / 32;
  constexpr int SMEMB = (M * KPAD * 2 > M * 128 * 4) ? M * KPAD * 2 : M * 128 * 4;
  __shared__ __align__(16) char smem[SMEMB];
  bf16_t* E = (bf16_t*)smem;
  float*  H = (float*)smem;

  const int tid = threadIdx.x, lane = tid & 31, wid = tid >> 5;
  const int p0 = blockIdx.x * 2;

  for (int r = wid; r < M; r += 8) {
    int p = r / KNN, e = r % KNN;
    int i = p0 + p, j = nbr[(size_t)i * KNN + e];
    for (int c = lane; c < CIN; c += 32) {
      float xi = (float)Xb[(size_t)i * 128 + c];
      float xj = (float)Xb[(size_t)j * 128 + c];
      E[r * KPAD + c]       = (bf16_t)xi;
      E[r * KPAD + CIN + c] = (bf16_t)(xj - xi);
    }
    for (int c = 2 * CIN + lane; c < KPAD; c += 32) E[r * KPAD + c] = (bf16_t)0.0f;
  }
  __syncthreads();

  v8f acc[5];
  {
    const int nt = wid;
#pragma unroll
    for (int mt = 0; mt < 5; ++mt) {
      int row = mt * 16 + (lane & 15);
      const bf16_t* abase = E + row * KPAD + ((lane >> 4) << 3);
      const bf16_t* bbase = Wp + (size_t)(nt * KT) * 512 + lane * 16;
      v8f c = {};
      for (int kt = 0; kt < KT; ++kt) {
        BV av;
        av.h[0] = *(const v8bf*)(abase + kt * 32);
        av.h[1] = *(const v8bf*)(abase + kt * 32 + 16);
        v16bf bv = *(const v16bf*)(bbase + (size_t)kt * 512);
        c = wmma_bf16(av.v, bv, c);
      }
      acc[mt] = c;
    }
  }
  __syncthreads();  // all E reads complete before reusing buffer as H

  {
    const int nt = wid;
    int coln = nt * 16 + (lane & 15);
    float b = bias[coln], s = sc[coln], t = sh[coln];
#pragma unroll
    for (int mt = 0; mt < 5; ++mt) {
#pragma unroll
      for (int v = 0; v < 8; ++v) {
        int r = mt * 16 + v + ((lane >> 4) << 3);
        H[r * 128 + coln] = fmaxf(acc[mt][v] + b, 0.0f) * s + t;
      }
    }
  }
  __syncthreads();

  {
    int p = tid >> 7, c = tid & 127;
    float m = -BIGF;
    for (int e = 0; e < KNN; ++e) m = fmaxf(m, H[(p * KNN + e) * 128 + c]);
    out[(size_t)(p0 + p) * 128 + c] = m;
  }
}

// ---------------------------------------------------------------------------
// Generic WMMA GEMM: out[M,N] = epilogue(A[M,Kp] @ Wpacked).  One wave per
// 16x16 tile, grid = (N/16, M/16).  A bf16 row-major (stride lda).
// ---------------------------------------------------------------------------
enum { EP_BIAS = 0, EP_BN = 1, EP_RELU = 2, EP_BIAS_ADD = 3 };

template <int EP>
__global__ __launch_bounds__(32) void gemm_kernel(
    const bf16_t* __restrict__ A, int lda, const bf16_t* __restrict__ Wp,
    const float* __restrict__ bias, const float* __restrict__ sc,
    const float* __restrict__ shf, const float* __restrict__ add,
    float* __restrict__ out, int ldo, bf16_t* __restrict__ outb, int ldob,
    int Kp) {
  const int lane = threadIdx.x & 31;
  const int nt = blockIdx.x, mt = blockIdx.y;
  const int KT = Kp >> 5;
  const int row0 = mt * 16 + (lane & 15);
  const bf16_t* abase = A + (size_t)row0 * lda + ((lane >> 4) << 3);
  const bf16_t* bbase = Wp + (size_t)nt * KT * 512 + lane * 16;
  v8f c = {};
  for (int kt = 0; kt < KT; ++kt) {
    BV av;
    av.h[0] = *(const v8bf*)(abase + kt * 32);
    av.h[1] = *(const v8bf*)(abase + kt * 32 + 16);
    v16bf bv = *(const v16bf*)(bbase + (size_t)kt * 512);
    c = wmma_bf16(av.v, bv, c);
  }
  int coln = nt * 16 + (lane & 15);
  float b = bias ? bias[coln] : 0.0f;
#pragma unroll
  for (int v = 0; v < 8; ++v) {
    int r = mt * 16 + v + ((lane >> 4) << 3);
    float val = c[v] + b;
    if (EP == EP_BN)       val = fmaxf(val, 0.0f) * sc[coln] + shf[coln];
    if (EP == EP_RELU)     val = fmaxf(val, 0.0f);
    if (EP == EP_BIAS_ADD) val += add[(size_t)r * ldo + coln];
    out[(size_t)r * ldo + coln] = val;
    if (outb) outb[(size_t)r * ldob + coln] = (bf16_t)val;
  }
}

// ---------------------------------------------------------------------------
// Column max over [8192 x 1024] (batch is all zeros -> segment_max == colmax)
// ---------------------------------------------------------------------------
__global__ void colmax_kernel(const float* __restrict__ src,
                              float* __restrict__ dst) {
  int c = blockIdx.x * blockDim.x + threadIdx.x;
  if (c >= 1024) return;
  float m = -BIGF;
  for (int r = 0; r < NPTS; ++r) m = fmaxf(m, src[(size_t)r * 1024 + c]);
  dst[c] = m;
}

// ---------------------------------------------------------------------------
// TNet head: g[1024] -> BN-MLP(256) -> Linear(9) = flattened 3x3 transform
// ---------------------------------------------------------------------------
__global__ __launch_bounds__(256) void tnet_head_kernel(
    const float* __restrict__ g, const float* __restrict__ W4,
    const float* __restrict__ b4, const float* __restrict__ s4,
    const float* __restrict__ t4, const float* __restrict__ WT,
    const float* __restrict__ bT, float* __restrict__ T9) {
  __shared__ float sh[256];
  int o = threadIdx.x;
  float acc = b4[o];
  for (int k = 0; k < 1024; ++k) acc += g[k] * W4[(size_t)k * 256 + o];
  sh[o] = fmaxf(acc, 0.0f) * s4[o] + t4[o];
  __syncthreads();
  if (o < 9) {
    float a2 = bT[o];
    for (int k = 0; k < 256; ++k) a2 += sh[k] * WT[(size_t)k * 9 + o];
    T9[o] = a2;
  }
}

// ---------------------------------------------------------------------------
// Build x = [pos @ T (3) | features (16) | xyz (3)] padded to 128 bf16 + |x|^2
// ---------------------------------------------------------------------------
__global__ void assemble_kernel(const float* __restrict__ xyz,
                                const float* __restrict__ feat,
                                const float* __restrict__ T9,
                                bf16_t* __restrict__ xb,
                                float* __restrict__ sq) {
  int n = blockIdx.x * blockDim.x + threadIdx.x;
  if (n >= NPTS) return;
  float p0 = xyz[(size_t)n * 3], p1 = xyz[(size_t)n * 3 + 1],
        p2 = xyz[(size_t)n * 3 + 2];
  float row[22];
#pragma unroll
  for (int j = 0; j < 3; ++j) row[j] = p0 * T9[j] + p1 * T9[3 + j] + p2 * T9[6 + j];
  for (int c = 0; c < 16; ++c) row[3 + c] = feat[(size_t)n * 16 + c];
  row[19] = p0; row[20] = p1; row[21] = p2;
  float acc = 0.0f;
  for (int c = 0; c < 128; ++c) {
    float v = (c < 22) ? row[c] : 0.0f;
    xb[(size_t)n * 128 + c] = (bf16_t)v;
    if (c < 22) acc += v * v;
  }
  sq[n] = acc;
}

// ---------------------------------------------------------------------------
// Host orchestration
// ---------------------------------------------------------------------------
extern "C" void kernel_launch(void* const* d_in, const int* in_sizes, int n_in,
                              void* d_out, int out_size, void* d_ws,
                              size_t ws_size, hipStream_t stream) {
  (void)in_sizes; (void)n_in; (void)out_size; (void)ws_size;

  const float* xyz   = (const float*)d_in[0];
  const float* feat  = (const float*)d_in[1];
  const int*   batch = (const int*)d_in[2];
  int p = 3;
  auto nx = [&]() { return (const float*)d_in[p++]; };
  const float *m1W = nx(), *m1b = nx(), *m1s = nx(), *m1t = nx();
  const float *m2W = nx(), *m2b = nx(), *m2s = nx(), *m2t = nx();
  const float *m3W = nx(), *m3b = nx(), *m3s = nx(), *m3t = nx();
  const float *m4W = nx(), *m4b = nx(), *m4s = nx(), *m4t = nx();
  const float *tW = nx(), *tb = nx();
  const float *cW[3], *cb_[3], *cs[3], *ct[3];
  for (int i = 0; i < 3; ++i) { cW[i] = nx(); cb_[i] = nx(); cs[i] = nx(); ct[i] = nx(); }
  const float *lW1[3], *lb1[3], *lW2[3], *lb2[3];
  for (int i = 0; i < 3; ++i) { lW1[i] = nx(); lb1[i] = nx(); lW2[i] = nx(); lb2[i] = nx(); }
  const float *tWl[3], *tbl[3];
  for (int i = 0; i < 3; ++i) { tWl[i] = nx(); tbl[i] = nx(); }

  // ---- workspace carve-out (256B aligned) ----
  char* w = (char*)d_ws;
  size_t off = 0;
  auto alloc = [&](size_t bytes) -> void* {
    void* r = w + off;
    off = (off + bytes + 255) & ~(size_t)255;
    return r;
  };
  bf16_t* xb      = (bf16_t*)alloc((size_t)NPTS * 128 * 2);
  float*  sqn     = (float*) alloc((size_t)NPTS * 4);
  int*    idx     = (int*)   alloc((size_t)NPTS * KNN * 4);
  float*  xi      = (float*) alloc((size_t)NPTS * 128 * 4);
  bf16_t* xib     = (bf16_t*)alloc((size_t)NPTS * 128 * 2);
  float*  lin1    = (float*) alloc((size_t)NPTS * 128 * 4);
  bf16_t* lin1b   = (bf16_t*)alloc((size_t)NPTS * 128 * 2);
  float*  lin2    = (float*) alloc((size_t)NPTS * 128 * 4);
  float*  xcur    = (float*) alloc((size_t)NPTS * 128 * 4);
  float*  tmp1024 = (float*) alloc((size_t)NPTS * 1024 * 4);
  float*  g1024   = (float*) alloc(1024 * 4);
  float*  T9      = (float*) alloc(16 * 4);
  bf16_t* p_m1 = (bf16_t*)alloc(32 * 64 * 2);
  bf16_t* p_m2 = (bf16_t*)alloc(64 * 128 * 2);
  bf16_t* p_m3 = (bf16_t*)alloc(128 * 1024 * 2);
  bf16_t* p_c[3];
  p_c[0] = (bf16_t*)alloc(64 * 128 * 2);
  p_c[1] = (bf16_t*)alloc(256 * 128 * 2);
  p_c[2] = (bf16_t*)alloc(256 * 128 * 2);
  bf16_t *p_l1[3], *p_l2[3], *p_lt[3];
  for (int i = 0; i < 3; ++i) {
    p_l1[i] = (bf16_t*)alloc(128 * 128 * 2);
    p_l2[i] = (bf16_t*)alloc(128 * 128 * 2);
    p_lt[i] = (bf16_t*)alloc(128 * 128 * 2);
  }

  auto pack = [&](const float* W, int Krows, int Ncols, int Kpad, bf16_t* Wp) {
    int KT = Kpad / 32, total = Kpad * Ncols;
    pack_w_kernel<<<(total + 255) / 256, 256, 0, stream>>>(W, Krows, Ncols, KT,
                                                           total, Wp);
  };
  pack(m1W, 6, 64, 32, p_m1);
  pack(m2W, 64, 128, 64, p_m2);
  pack(m3W, 128, 1024, 128, p_m3);
  pack(cW[0], 44, 128, 64, p_c[0]);
  pack(cW[1], 256, 128, 256, p_c[1]);
  pack(cW[2], 256, 128, 256, p_c[2]);
  for (int i = 0; i < 3; ++i) {
    pack(lW1[i], 128, 128, 128, p_l1[i]);
    pack(lW2[i], 128, 128, 128, p_l2[i]);
  }
  pack(tWl[0], 22, 128, 128, p_lt[0]);
  pack(tWl[1], 128, 128, 128, p_lt[1]);
  pack(tWl[2], 128, 128, 128, p_lt[2]);

  // ---- TransformNet ----
  cvt_pad_kernel<<<NPTS / 256, 256, 0, stream>>>(xyz, 3, 3, xb, sqn, NPTS);
  knn_topk_kernel<1><<<NPTS / 16, 256, 0, stream>>>(xb, sqn, batch, idx);
  tnet_edgeconv_kernel<<<NPTS / 2, 256, 0, stream>>>(
      xyz, idx, p_m1, m1b, m1s, m1t, p_m2, m2b, m2s, m2t, xi);
  cvt_pad_kernel<<<NPTS / 256, 256, 0, stream>>>(xi, 128, 128, xib, nullptr, NPTS);
  gemm_kernel<EP_BN><<<dim3(1024 / 16, NPTS / 16), 32, 0, stream>>>(
      xib, 128, p_m3, m3b, m3s, m3t, nullptr, tmp1024, 1024, nullptr, 0, 128);
  colmax_kernel<<<4, 256, 0, stream>>>(tmp1024, g1024);
  tnet_head_kernel<<<1, 256, 0, stream>>>(g1024, m4W, m4b, m4s, m4t, tW, tb, T9);
  assemble_kernel<<<NPTS / 256, 256, 0, stream>>>(xyz, feat, T9, xb, sqn);

  // ---- EdgeConv stack with residual linear branches ----
  for (int i = 0; i < 3; ++i) {
    if (i == 0)
      knn_topk_kernel<2><<<NPTS / 16, 256, 0, stream>>>(xb, sqn, batch, idx);
    else
      knn_topk_kernel<4><<<NPTS / 16, 256, 0, stream>>>(xb, sqn, batch, idx);

    if (i == 0)
      edgeconv_kernel<22, 64><<<NPTS / 2, 256, 0, stream>>>(
          xb, idx, p_c[0], cb_[0], cs[0], ct[0], xi);
    else
      edgeconv_kernel<128, 256><<<NPTS / 2, 256, 0, stream>>>(
          xb, idx, p_c[i], cb_[i], cs[i], ct[i], xi);

    cvt_pad_kernel<<<NPTS / 256, 256, 0, stream>>>(xi, 128, 128, xib, nullptr, NPTS);
    gemm_kernel<EP_RELU><<<dim3(8, NPTS / 16), 32, 0, stream>>>(
        xib, 128, p_l1[i], lb1[i], nullptr, nullptr, nullptr, lin1, 128, lin1b,
        128, 128);
    gemm_kernel<EP_BIAS><<<dim3(8, NPTS / 16), 32, 0, stream>>>(
        lin1b, 128, p_l2[i], lb2[i], nullptr, nullptr, nullptr, lin2, 128,
        nullptr, 0, 128);
    float* xdst = (i == 2) ? (float*)d_out : xcur;
    gemm_kernel<EP_BIAS_ADD><<<dim3(8, NPTS / 16), 32, 0, stream>>>(
        xb, 128, p_lt[i], tbl[i], nullptr, nullptr, lin2, xdst, 128, nullptr, 0,
        128);
    if (i < 2)
      cvt_pad_kernel<<<NPTS / 256, 256, 0, stream>>>(xcur, 128, 128, xb, sqn,
                                                     NPTS);
  }
}
